// DSAttention_51153060496044
// MI455X (gfx1250) — matrix operand order repaired and checked by reading, not compiled, not run on hardware
//
#include <hip/hip_runtime.h>

typedef __attribute__((ext_vector_type(16))) _Float16 v16h;
typedef __attribute__((ext_vector_type(8)))  _Float16 v8h;
typedef __attribute__((ext_vector_type(8)))  float    v8f;

#define WMMA_F16(a, b, c) \
  __builtin_amdgcn_wmma_f32_16x16x32_f16(false, (a), false, (b), (short)0, (c), false, false)

constexpr int Bc = 4, Lc = 2048, Sc = 2048, Hc = 8, Ec = 64;
constexpr int QTILE = 64;   // query rows per block (4 waves x 16)
constexpr int KSTEP = 32;   // keys per S-iteration

__global__ __launch_bounds__(128) void dsattn_fa(
    const float* __restrict__ Q, const float* __restrict__ K,
    const float* __restrict__ V, const float* __restrict__ TAU,
    const float* __restrict__ DELTA, float* __restrict__ OUT)
{
  // Double-buffered raw fp32 tiles, filled by GLOBAL_LOAD_ASYNC_TO_LDS (no VGPR round trip)
  __shared__ __align__(16) float rawK[2][KSTEP * Ec];    // 2 x 8 KB
  __shared__ __align__(16) float rawV[2][KSTEP * Ec];    // 2 x 8 KB
  // K tile transposed [e][s] (f16) -> contiguous B-operand reads for QK^T
  __shared__ __align__(32) _Float16 kT[Ec * KSTEP];      // 4 KB
  // V tile row-major [s][e] (f16) -> contiguous B-operand reads for P*V
  __shared__ __align__(32) _Float16 vS[KSTEP * Ec];      // 4 KB
  // per-wave P staging (C-layout f32 -> A-layout f16 round trip)
  __shared__ __align__(32) _Float16 pT[4][16][32];       // 4 KB

  const int tid  = threadIdx.x;
  const int wave = tid >> 5;
  const int lane = tid & 31;
  const int l16  = lane & 15;
  const int hi   = lane >> 4;     // half-wave select
  const int hi8  = hi * 8;

  const int nq   = Lc / QTILE;    // 32
  const int bid  = blockIdx.x;
  const int qblk = bid % nq;
  const int h    = (bid / nq) % Hc;
  const int b    = bid / (nq * Hc);
  const int q0   = qblk * QTILE;
  const int qbase = q0 + wave * 16;

  const float tscale  = TAU[b] * 0.125f;           // tau * (1/sqrt(E))
  const long  strideS = (long)Hc * Ec;             // stride between l/s rows
  const float* Qb = Q + (((long)b * Lc) * Hc + h) * Ec;
  const float* Kb = K + (((long)b * Sc) * Hc + h) * Ec;
  const float* Vb = V + (((long)b * Sc) * Hc + h) * Ec;
  const float* Db = DELTA + (long)b * Sc;
  float*       Ob = OUT + (((long)b * Lc) * Hc + h) * Ec;

  // Issue async global->LDS copies for the K/V tile starting at s0 into buffer `buf`.
  // 512 x 16B chunks per 8KB tile; each of 128 threads issues 4 chunks per tensor.
  auto issueTile = [&](int bufSel, int s0) {
    #pragma unroll
    for (int j = 0; j < 4; ++j) {
      const int c    = j * 128 + tid;        // chunk id 0..511
      const int srel = c >> 4;               // tile row
      const int eoff = (c & 15) * 4;         // float offset within row
      const float* kg = Kb + (long)(s0 + srel) * strideS + eoff;
      const float* vg = Vb + (long)(s0 + srel) * strideS + eoff;
      const unsigned kd = (unsigned)(uintptr_t)&rawK[bufSel][c * 4];
      const unsigned vd = (unsigned)(uintptr_t)&rawV[bufSel][c * 4];
      asm volatile("global_load_async_to_lds_b128 %0, %1, off"
                   :: "v"(kd), "v"((unsigned long long)(uintptr_t)kg) : "memory");
      asm volatile("global_load_async_to_lds_b128 %0, %1, off"
                   :: "v"(vd), "v"((unsigned long long)(uintptr_t)vg) : "memory");
    }
  };

  // ---- Q tile as two A-operands (E split 0-31 / 32-63), kept in VGPRs ----
  v16h qa[2];
  {
    const float* qp = Qb + (long)(qbase + l16) * strideS;
    #pragma unroll
    for (int c = 0; c < 2; ++c) {
      const int e1 = c * 32 + hi8;        // K elements 0..7 of operand
      const int e2 = e1 + 16;             // K elements 8..15 of operand
      v16h a;
      #pragma unroll
      for (int i = 0; i < 8; ++i) {
        a[i]     = (_Float16)qp[e1 + i];
        a[i + 8] = (_Float16)qp[e2 + i];
      }
      qa[c] = a;
    }
  }

  v8f o0 = {}, o1 = {}, o2 = {}, o3 = {};
  float mi[8], li[8];
  #pragma unroll
  for (int r = 0; r < 8; ++r) { mi[r] = -3.0e38f; li[r] = 0.0f; }

  const int sEnd = q0 + QTILE;            // causal frontier for this block
  int buf = 0;
  issueTile(0, 0);                        // prologue: async-prefetch tile 0

  for (int s0 = 0; s0 < sEnd; s0 += KSTEP) {
    asm volatile("s_wait_asynccnt 0" ::: "memory");  // my tile chunks landed in LDS
    __syncthreads();   // all waves' async done + prev compute done with kT/vS

    // Overlap: kick off next tile's async copies into the alternate buffer
    if (s0 + KSTEP < sEnd) issueTile(buf ^ 1, s0 + KSTEP);

    // ---- LDS->LDS convert: raw fp32 -> kT (transposed f16) / vS (f16) ----
    {
      const int srel  = tid & 31;
      const int ebase = (tid >> 5) * 16;
      const float* kr = &rawK[buf][srel * Ec + ebase];
      const float* vr = &rawV[buf][srel * Ec + ebase];
      #pragma unroll
      for (int i = 0; i < 16; ++i)
        kT[(ebase + i) * KSTEP + srel] = (_Float16)kr[i];
      v16h vv;
      #pragma unroll
      for (int i = 0; i < 16; ++i) vv[i] = (_Float16)vr[i];
      *(v16h*)&vS[srel * Ec + ebase] = vv;
    }
    __syncthreads();

    // ---- QK^T: two 16-key subtiles, each accumulating over E in 2 chunks ----
    v8f acc[2];
    acc[0] = v8f{}; acc[1] = v8f{};
    #pragma unroll
    for (int t = 0; t < 2; ++t) {
      #pragma unroll
      for (int c = 0; c < 2; ++c) {
        const int e = c * 32 + hi8;                         // B elements = K-dim (e)
        v16h kb;                                            // lane = N (s)
        #pragma unroll
        for (int i = 0; i < 8; ++i) {
          kb[i]     = kT[(e + i) * KSTEP + t * 16 + l16];
          kb[i + 8] = kT[(e + 16 + i) * KSTEP + t * 16 + l16];
        }
        acc[t] = WMMA_F16(qa[c], kb, acc[t]);
      }
    }

    // ---- logits + causal mask + online softmax ----
    const float d0 = Db[s0 + l16] * 0.125f;
    const float d1 = Db[s0 + 16 + l16] * 0.125f;
    const int sg0 = s0 + l16;
    const int sg1 = s0 + 16 + l16;
    float p0[8], p1[8], corr[8];
    #pragma unroll
    for (int r = 0; r < 8; ++r) {
      const int qrow = qbase + r + hi8;                     // C-layout row
      float x0 = (sg0 <= qrow) ? (acc[0][r] * tscale + d0) : -3.0e38f;
      float x1 = (sg1 <= qrow) ? (acc[1][r] * tscale + d1) : -3.0e38f;
      float mx = fmaxf(x0, x1);
      #pragma unroll
      for (int off = 1; off < 16; off <<= 1)
        mx = fmaxf(mx, __shfl_xor(mx, off, 32));
      const float mnew = fmaxf(mi[r], mx);
      const float e0 = __expf(x0 - mnew);
      const float e1 = __expf(x1 - mnew);
      float rs = e0 + e1;
      #pragma unroll
      for (int off = 1; off < 16; off <<= 1)
        rs += __shfl_xor(rs, off, 32);
      const float cr = __expf(mi[r] - mnew);
      li[r] = li[r] * cr + rs;
      mi[r] = mnew;
      corr[r] = cr;
      p0[r] = e0; p1[r] = e1;
    }

    // ---- rescale O, emit P (C-layout -> LDS) ----
    #pragma unroll
    for (int r = 0; r < 8; ++r) {
      o0[r] *= corr[r]; o1[r] *= corr[r]; o2[r] *= corr[r]; o3[r] *= corr[r];
      pT[wave][r + hi8][l16]      = (_Float16)p0[r];
      pT[wave][r + hi8][16 + l16] = (_Float16)p1[r];
    }
    asm volatile("s_wait_dscnt 0" ::: "memory");   // wave-local LDS RAW fence

    // ---- reload P as A-operand (lane = M, element runs per half-wave) ----
    v16h pa;
    {
      const _Float16* pr = &pT[wave][l16][0];
      v8h a0 = *(const v8h*)(pr + hi8);
      v8h a1 = *(const v8h*)(pr + hi8 + 16);
      #pragma unroll
      for (int i = 0; i < 8; ++i) { pa[i] = a0[i]; pa[i + 8] = a1[i]; }
    }

    // ---- O += P * V  (four 16-wide E chunks) ----
    v16h vb0, vb1, vb2, vb3;
    #pragma unroll
    for (int i = 0; i < 8; ++i) {
      const int k0 = (hi8 + i) * Ec + l16, k1 = (hi8 + 16 + i) * Ec + l16;
      vb0[i] = vS[k0 +  0]; vb0[i + 8] = vS[k1 +  0];
      vb1[i] = vS[k0 + 16]; vb1[i + 8] = vS[k1 + 16];
      vb2[i] = vS[k0 + 32]; vb2[i + 8] = vS[k1 + 32];
      vb3[i] = vS[k0 + 48]; vb3[i + 8] = vS[k1 + 48];
    }
    o0 = WMMA_F16(pa, vb0, o0);
    o1 = WMMA_F16(pa, vb1, o1);
    o2 = WMMA_F16(pa, vb2, o2);
    o3 = WMMA_F16(pa, vb3, o3);

    buf ^= 1;
  }

  // ---- normalize and store (C layout: VGPR r -> row r / r+8) ----
  #pragma unroll
  for (int r = 0; r < 8; ++r) {
    const float inv = 1.0f / li[r];
    float* op = Ob + (long)(qbase + r + hi8) * strideS;
    op[l16]      = o0[r] * inv;
    op[l16 + 16] = o1[r] * inv;
    op[l16 + 32] = o2[r] * inv;
    op[l16 + 48] = o3[r] * inv;
  }
}

extern "C" void kernel_launch(void* const* d_in, const int* in_sizes, int n_in,
                              void* d_out, int out_size, void* d_ws, size_t ws_size,
                              hipStream_t stream) {
  (void)in_sizes; (void)n_in; (void)out_size; (void)d_ws; (void)ws_size;
  const float* q     = (const float*)d_in[0];
  const float* k     = (const float*)d_in[1];
  const float* v     = (const float*)d_in[2];
  const float* tau   = (const float*)d_in[3];
  const float* delta = (const float*)d_in[4];
  float* out = (float*)d_out;

  const int grid = Bc * Hc * (Lc / QTILE);   // 1024 blocks of 128 threads
  dsattn_fa<<<grid, 128, 0, stream>>>(q, k, v, tau, delta, out);
}
